// IntraPositionAttention_11776800325696
// MI455X (gfx1250) — matrix operand order repaired and checked by reading, not compiled, not run on hardware
//
#include <hip/hip_runtime.h>

// ---------------------------------------------------------------------------
// IntraPositionAttention for MI455X (gfx1250, wave32, WMMA).
//   * 4 big GEMMs (275 GFLOP) on v_wmma_f32_16x16x32_bf16, fp32 accumulate
//   * fp32 -> bf16 conversion as packed-u32 passes (no half-reg VALU noise)
//   * GEMM tiles move global->LDS via GLOBAL_LOAD_ASYNC_TO_LDS_B128 with
//     double buffering (ASYNCcnt), falling back to register staging if the
//     builtin is unavailable
//   * tiny 8x8-per-head attention on VALU+LDS (not WMMA-worthy: M=8<16)
// ---------------------------------------------------------------------------

typedef __bf16 v16bf __attribute__((ext_vector_type(16)));
typedef __bf16 v8bf  __attribute__((ext_vector_type(8)));
typedef float  v8f   __attribute__((ext_vector_type(8)));
typedef int    v4i   __attribute__((ext_vector_type(4)));

#define DMODEL 1024
#define MROWS  32768   // B*L*K = 2*2048*8
#define NHEADS 16
#define HDIM   64
#define KSLOT  8

#define GLOBAL_AS __attribute__((address_space(1)))
#define LDS_AS    __attribute__((address_space(3)))

#if defined(__HIP_DEVICE_COMPILE__) && __has_builtin(__builtin_amdgcn_global_load_async_to_lds_b128)
#define HAVE_ASYNC 1
#else
#define HAVE_ASYNC 0
#endif

// ---- packed bf16 helpers (all 32-bit integer ALU, no scalar __bf16) -------
static __device__ __forceinline__ unsigned pack_bf16(float f0, float f1) {
    unsigned a = __builtin_bit_cast(unsigned, f0);
    unsigned b = __builtin_bit_cast(unsigned, f1);
    a = (a + 0x7FFFu + ((a >> 16) & 1u)) >> 16;          // rne, low half
    b = (b + 0x7FFFu + ((b >> 16) & 1u)) & 0xFFFF0000u;  // rne, high half
    return a | b;
}
static __device__ __forceinline__ float u2f_lo(unsigned u) {
    return __builtin_bit_cast(float, u << 16);
}
static __device__ __forceinline__ float u2f_hi(unsigned u) {
    return __builtin_bit_cast(float, u & 0xFFFF0000u);
}
static __device__ __forceinline__ __bf16 f2bf(float f) {
    unsigned u = __builtin_bit_cast(unsigned, f);
    unsigned short h = (unsigned short)((u + 0x7FFFu + ((u >> 16) & 1u)) >> 16);
    return __builtin_bit_cast(__bf16, h);
}

static __device__ __forceinline__ void wait_async0() {
#if HAVE_ASYNC
#if __has_builtin(__builtin_amdgcn_s_wait_asynccnt)
    __builtin_amdgcn_s_wait_asynccnt(0);
#else
    asm volatile("s_wait_asynccnt 0x0" ::: "memory");
#endif
#endif
}

#if HAVE_ASYNC
// async 16-byte global->LDS copy; builtin expects v4i pointers (b128 payload)
static __device__ __forceinline__ void async_b128(const void* gsrc, void* lds) {
    __builtin_amdgcn_global_load_async_to_lds_b128(
        (GLOBAL_AS v4i*)(GLOBAL_AS void*)gsrc,
        (LDS_AS v4i*)(LDS_AS void*)lds, 0, 0);
}
#endif

// ---------------------------------------------------------------------------
// fp32 -> bf16 elementwise (8 elems/thread, packed u32 math, uint4 stores)
// ---------------------------------------------------------------------------
__global__ __launch_bounds__(256) void cvt_f32_bf16(
    const float* __restrict__ src, unsigned* __restrict__ dst4 /* = bf16*2 */,
    int n8)
{
    int idx = blockIdx.x * 256 + threadIdx.x;
    if (idx >= n8) return;
    const float4 a = *(const float4*)(src + (size_t)idx * 8);
    const float4 b = *(const float4*)(src + (size_t)idx * 8 + 4);
    uint4 r;
    r.x = pack_bf16(a.x, a.y);
    r.y = pack_bf16(a.z, a.w);
    r.z = pack_bf16(b.x, b.y);
    r.w = pack_bf16(b.z, b.w);
    *(uint4*)(dst4 + (size_t)idx * 4) = r;
}

// ---------------------------------------------------------------------------
// Tiled WMMA GEMM:  Out[z][m][n] = A[m][:] . W[z][n][:] + bias_z[n]
// Tile 128x128x32, 256 threads = 8 waves; wave owns 32x64 = 2x4 WMMA accums.
// Double-buffered LDS; tiles DMA'd with GLOBAL_LOAD_ASYNC_TO_LDS_B128.
// ---------------------------------------------------------------------------
template <typename OutT>
__global__ __launch_bounds__(256) void gemm_wmma(
    const __bf16* __restrict__ A,     // [MROWS][1024] bf16
    const __bf16* __restrict__ Wall,  // [gridDim.z][1024][1024] bf16
    const float* __restrict__ b0, const float* __restrict__ b1,
    const float* __restrict__ b2,
    OutT* __restrict__ Out)           // [gridDim.z][MROWS][1024]
{
    const int z = blockIdx.z;
    const __bf16* W = Wall + (size_t)z * DMODEL * DMODEL;
    const float* bias = (z == 0) ? b0 : (z == 1 ? b1 : b2);
    OutT* out = Out + (size_t)z * MROWS * DMODEL;

    const int n0 = blockIdx.x * 128;
    const int m0 = blockIdx.y * 128;
    const int tid  = threadIdx.x;
    const int lane = tid & 31;
    const int wave = tid >> 5;
    const int waveM = wave & 3;   // 4 waves along M (32 rows each)
    const int waveN = wave >> 2;  // 2 waves along N (64 cols each)

    __shared__ __attribute__((aligned(16))) __bf16 As[2 * 128 * 32];
    __shared__ __attribute__((aligned(16))) __bf16 Bs[2 * 128 * 32];

    v8f acc[2][4] = {};

    // cooperative copy: tile = 512 x 16B chunks; thread owns chunks tid, tid+256
    const int car0 = tid >> 2;            // rows 0..63
    const int cac  = (tid & 3) * 8;       // k-chunk: 0,8,16,24
    const int car1 = car0 + 64;           // rows 64..127
    const __bf16* Abase = A + (size_t)m0 * DMODEL;
    const __bf16* Bbase = W + (size_t)n0 * DMODEL;

    const int fr = lane & 15;             // fragment row/col within 16
    const int fg = lane >> 4;             // lane group (K-half select)

#if HAVE_ASYNC
    auto issue_tile = [&](int buf, int k0) {
        const int lo = buf * (128 * 32);
        async_b128(Abase + (size_t)car0 * DMODEL + k0 + cac,
                   As + lo + car0 * 32 + cac);
        async_b128(Abase + (size_t)car1 * DMODEL + k0 + cac,
                   As + lo + car1 * 32 + cac);
        async_b128(Bbase + (size_t)car0 * DMODEL + k0 + cac,
                   Bs + lo + car0 * 32 + cac);
        async_b128(Bbase + (size_t)car1 * DMODEL + k0 + cac,
                   Bs + lo + car1 * 32 + cac);
    };
    issue_tile(0, 0);
    for (int kt = 0; kt < DMODEL / 32; ++kt) {
        const int cur = kt & 1;
        wait_async0();          // my async copies into `cur` landed
        __syncthreads();        // everyone's landed; prev compute finished
        if (kt + 1 < DMODEL / 32) issue_tile(cur ^ 1, (kt + 1) * 32);
#else
    for (int kt = 0; kt < DMODEL / 32; ++kt) {
        const int cur = 0;
        const int k0 = kt * 32;
        uint4 ra0 = *(const uint4*)(Abase + (size_t)car0 * DMODEL + k0 + cac);
        uint4 ra1 = *(const uint4*)(Abase + (size_t)car1 * DMODEL + k0 + cac);
        uint4 rb0 = *(const uint4*)(Bbase + (size_t)car0 * DMODEL + k0 + cac);
        uint4 rb1 = *(const uint4*)(Bbase + (size_t)car1 * DMODEL + k0 + cac);
        __syncthreads();        // previous compute done before overwrite
        *(uint4*)&As[car0 * 32 + cac] = ra0;
        *(uint4*)&As[car1 * 32 + cac] = ra1;
        *(uint4*)&Bs[car0 * 32 + cac] = rb0;
        *(uint4*)&Bs[car1 * 32 + cac] = rb1;
        __syncthreads();
#endif
        // ---- fragments per ISA 7.12.2 layouts ----
        const __bf16* abuf = As + cur * (128 * 32);
        const __bf16* bbuf = Bs + cur * (128 * 32);
        v16bf af[2], bfv[4];
        #pragma unroll
        for (int mt = 0; mt < 2; ++mt) {
            // A 16x32: lane holds row fr, K=[fg*8,fg*8+8) then [16+fg*8,...)
            const __bf16* p = abuf + (waveM * 32 + mt * 16 + fr) * 32;
            v8bf lo = *(const v8bf*)(p + fg * 8);
            v8bf hi = *(const v8bf*)(p + 16 + fg * 8);
            af[mt] = __builtin_shufflevector(lo, hi,
                     0,1,2,3,4,5,6,7,8,9,10,11,12,13,14,15);
        }
        #pragma unroll
        for (int nt = 0; nt < 4; ++nt) {
            // B 32x16: lane holds col n=fr, K=[fg*8,fg*8+8) then [16+fg*8,...)
            const __bf16* p = bbuf + (waveN * 64 + nt * 16 + fr) * 32;
            v8bf lo = *(const v8bf*)(p + fg * 8);
            v8bf hi = *(const v8bf*)(p + 16 + fg * 8);
            bfv[nt] = __builtin_shufflevector(lo, hi,
                      0,1,2,3,4,5,6,7,8,9,10,11,12,13,14,15);
        }
        #pragma unroll
        for (int mt = 0; mt < 2; ++mt)
            #pragma unroll
            for (int nt = 0; nt < 4; ++nt)
                acc[mt][nt] = __builtin_amdgcn_wmma_f32_16x16x32_bf16(
                    false, af[mt], false, bfv[nt],
                    (short)0, acc[mt][nt], false, false);
#if !HAVE_ASYNC
        __syncthreads();
#endif
    }

    // ---- epilogue: D elem i of VGPR -> row fg*8+i, col lane%16 ----
    #pragma unroll
    for (int nt = 0; nt < 4; ++nt) {
        const int col = n0 + waveN * 64 + nt * 16 + fr;
        const float bv = bias[col];
        #pragma unroll
        for (int mt = 0; mt < 2; ++mt) {
            #pragma unroll
            for (int i = 0; i < 8; ++i) {
                const int row = m0 + waveM * 32 + mt * 16 + fg * 8 + i;
                float vres = acc[mt][nt][i] + bv;
                if constexpr (sizeof(OutT) == 4)
                    ((float*)out)[(size_t)row * DMODEL + col] = vres;
                else
                    ((__bf16*)out)[(size_t)row * DMODEL + col] = f2bf(vres);
            }
        }
    }
}

// ---------------------------------------------------------------------------
// Per-position attention: one 256-thread block per (b,l).
// 16 heads x (8x8 over hd=64) + softmax + PV, bf16 handled as packed u32.
// ---------------------------------------------------------------------------
__global__ __launch_bounds__(256) void attn_kernel(
    const __bf16* __restrict__ qkv,   // [3][MROWS][1024] bf16
    const float* __restrict__ probs,  // [B*L*K]
    const float* __restrict__ pscale_ptr,
    __bf16* __restrict__ attn_out)    // [MROWS][1024] bf16
{
    __shared__ __attribute__((aligned(16))) __bf16 s[3 * KSLOT * DMODEL]; // 48KB
    __shared__ float sc[NHEADS * KSLOT * KSLOT];                          // 4KB
    __shared__ float pb[KSLOT];

    const int bl = blockIdx.x;                 // 0 .. B*L-1
    const size_t rowbase = (size_t)bl * KSLOT;
    const int tid = threadIdx.x;

    // stage q,k,v: 3*8*1024 bf16 in 16B chunks
    for (int c = tid; c < 3 * KSLOT * (DMODEL / 8); c += 256) {
        int mat = c / (KSLOT * (DMODEL / 8));
        int rem = c - mat * (KSLOT * (DMODEL / 8));
        int row = rem / (DMODEL / 8);
        int ch  = rem % (DMODEL / 8);
        const __bf16* src = qkv + (size_t)mat * MROWS * DMODEL
                          + (rowbase + row) * DMODEL + ch * 8;
        *(uint4*)&s[mat * KSLOT * DMODEL + row * DMODEL + ch * 8] =
            *(const uint4*)src;
    }
    if (tid < KSLOT)
        pb[tid] = pscale_ptr[0] * __logf(probs[rowbase + tid] + 1e-8f);
    __syncthreads();

    // scores: 16 heads * 8 * 8 = 1024 entries (4/thread), u32-pair dot
    for (int e = tid; e < NHEADS * KSLOT * KSLOT; e += 256) {
        int h = e >> 6; int i = (e >> 3) & 7; int j = e & 7;
        const unsigned* qp = (const unsigned*)&s[i * DMODEL + h * HDIM];
        const unsigned* kp = (const unsigned*)&s[KSLOT * DMODEL + j * DMODEL + h * HDIM];
        float d = 0.f;
        #pragma unroll
        for (int t = 0; t < HDIM / 2; ++t) {
            unsigned qu = qp[t], ku = kp[t];
            d = fmaf(u2f_lo(qu), u2f_lo(ku), d);
            d = fmaf(u2f_hi(qu), u2f_hi(ku), d);
        }
        sc[e] = d * 0.125f + pb[j];   // scale = 1/sqrt(64)
    }
    __syncthreads();

    // softmax over j for each (h,i): 128 rows
    if (tid < NHEADS * KSLOT) {
        float* r = &sc[tid * KSLOT];
        float m = r[0];
        #pragma unroll
        for (int j = 1; j < KSLOT; ++j) m = fmaxf(m, r[j]);
        float sum = 0.f;
        #pragma unroll
        for (int j = 0; j < KSLOT; ++j) { float e = __expf(r[j] - m); r[j] = e; sum += e; }
        float inv = 1.f / sum;
        #pragma unroll
        for (int j = 0; j < KSLOT; ++j) r[j] *= inv;
    }
    __syncthreads();

    // out[h,i,d] = sum_j P[h,i,j] * v[j][h*64+d]; two d's per thread-step,
    // written as one packed-u32 store of 2 bf16.
    for (int e = tid; e < NHEADS * KSLOT * (HDIM / 2); e += 256) {
        int h = e >> 8; int i = (e >> 5) & 7; int d2 = e & 31;
        const float* p = &sc[(h * KSLOT + i) * KSLOT];
        const unsigned* vbase =
            (const unsigned*)&s[2 * KSLOT * DMODEL + h * HDIM + d2 * 2];
        float o0 = 0.f, o1 = 0.f;
        #pragma unroll
        for (int j = 0; j < KSLOT; ++j) {
            unsigned vu = vbase[j * (DMODEL / 2)];
            o0 = fmaf(p[j], u2f_lo(vu), o0);
            o1 = fmaf(p[j], u2f_hi(vu), o1);
        }
        unsigned* dst = (unsigned*)(attn_out + (rowbase + i) * DMODEL + h * HDIM + d2 * 2);
        *dst = pack_bf16(o0, o1);
    }
}

// ---------------------------------------------------------------------------
// Launch
// ---------------------------------------------------------------------------
extern "C" void kernel_launch(void* const* d_in, const int* in_sizes, int n_in,
                              void* d_out, int out_size, void* d_ws, size_t ws_size,
                              hipStream_t stream) {
    const float* h      = (const float*)d_in[0];
    const float* probs  = (const float*)d_in[1];
    const float* Wq     = (const float*)d_in[2];
    const float* bq     = (const float*)d_in[3];
    const float* Wk     = (const float*)d_in[4];
    const float* bk     = (const float*)d_in[5];
    const float* Wv     = (const float*)d_in[6];
    const float* bv     = (const float*)d_in[7];
    const float* Wo     = (const float*)d_in[8];
    const float* bo     = (const float*)d_in[9];
    const float* pscale = (const float*)d_in[10];
    float* out = (float*)d_out;

    // workspace layout (328 MB total)
    char* ws = (char*)d_ws;
    __bf16* wbf  = (__bf16*)ws;                                   //   8 MB [4][1024][1024]
    __bf16* hbf  = (__bf16*)(ws + (size_t)8   * 1024 * 1024);     //  64 MB [M][1024]
    __bf16* qkv  = (__bf16*)(ws + (size_t)72  * 1024 * 1024);     // 192 MB [3][M][1024]
    __bf16* attn = (__bf16*)(ws + (size_t)264 * 1024 * 1024);     //  64 MB [M][1024]

    const size_t WELEM = (size_t)DMODEL * DMODEL;  // 1M elems per weight

    // 1) fp32 -> bf16: weights (4 x 1M) and h (32M), packed-u32 conversion
    cvt_f32_bf16<<<512, 256, 0, stream>>>(Wq, (unsigned*)(wbf + 0 * WELEM), (int)(WELEM / 8));
    cvt_f32_bf16<<<512, 256, 0, stream>>>(Wk, (unsigned*)(wbf + 1 * WELEM), (int)(WELEM / 8));
    cvt_f32_bf16<<<512, 256, 0, stream>>>(Wv, (unsigned*)(wbf + 2 * WELEM), (int)(WELEM / 8));
    cvt_f32_bf16<<<512, 256, 0, stream>>>(Wo, (unsigned*)(wbf + 3 * WELEM), (int)(WELEM / 8));
    cvt_f32_bf16<<<16384, 256, 0, stream>>>(h, (unsigned*)hbf, (int)((size_t)MROWS * DMODEL / 8));

    // 2) fused Q/K/V projections: grid.z selects matrix; M/128=256, N/128=8
    gemm_wmma<__bf16><<<dim3(8, 256, 3), 256, 0, stream>>>(
        hbf, wbf, bq, bk, bv, qkv);

    // 3) per-position multi-head attention (B*L = 4096 blocks)
    attn_kernel<<<4096, 256, 0, stream>>>(qkv, probs, pscale, attn);

    // 4) output projection -> fp32 d_out
    gemm_wmma<float><<<dim3(8, 256, 1), 256, 0, stream>>>(
        attn, wbf + 3 * WELEM, bo, bo, bo, out);
}
